// LatentSDE_56796647522528
// MI455X (gfx1250) — matrix operand order repaired and checked by reading, not compiled, not run on hardware
//
#include <hip/hip_runtime.h>
#include <hip/hip_bf16.h>
#include <math.h>

typedef __bf16 bf16;
typedef unsigned int u32;
typedef __attribute__((ext_vector_type(2)))  u32    u32x2;
typedef __attribute__((ext_vector_type(4)))  u32    u32x4;
typedef __attribute__((ext_vector_type(8)))  int    i32x8;
typedef __attribute__((ext_vector_type(4)))  int    i32x4;
typedef __attribute__((ext_vector_type(16))) __bf16 v16bf;
typedef __attribute__((ext_vector_type(8)))  float  v8f;

namespace {
constexpr int T_ = 128, B_ = 256, D_ = 128, L_ = 32, C_ = 64, H_ = 256, S_ = 16;
constexpr int SB_ = S_ * B_;                 // 4096
constexpr float DT_ = 0.02f;
}

// fast softplus / sigmoid on the hardware transcendental pipe
__device__ __forceinline__ float splus(float x) {
    return fmaxf(x, 0.f) + __logf(1.f + __expf(-fabsf(x)));
}
__device__ __forceinline__ float sigm(float x) { return 1.f / (1.f + __expf(-x)); }

union FragBf { v16bf v; u32x4 q[2]; };

// ---------------------------------------------------------------------------
// Stage a 64 x K bf16 weight panel (row-major, row stride K elements) into
// LDS. Preferred path: Tensor Data Mover (gfx1250 TDM) -- one DMA descriptor
// issued by wave 0, completion via s_wait_tensorcnt. Rows past N are OOB in
// the tensor descriptor and return zero. Fallback: cooperative vector loads.
// This toolchain uses the 6-arg builtin: (g0,u4)(g1,i8)(g2,i4)(g3,i4)(g4,i8)(cpol).
// ---------------------------------------------------------------------------
__device__ __forceinline__ void stage_w_panel(const bf16* W, int N, int K, int bn,
                                              bf16* Ws, int tid) {
#if defined(__gfx1250__) && __has_builtin(__builtin_amdgcn_tensor_load_to_lds) && \
    __has_builtin(__builtin_amdgcn_s_wait_tensorcnt)
    if (tid < 32) {   // one wave issues the DMA (EXEC ignored by TDM)
        const unsigned long long ga =
            (unsigned long long)(uintptr_t)(W + (size_t)bn * K);
        const unsigned lds_off = (unsigned)(uintptr_t)Ws;   // low 32b = LDS offset
        int rows = N - bn; if (rows > 64) rows = 64;
        u32x4 g0; i32x8 g1;
        i32x4 g2 = {0, 0, 0, 0}; i32x4 g3 = {0, 0, 0, 0};
        i32x8 g4 = {0, 0, 0, 0, 0, 0, 0, 0};
        // D# group0: count=1 | lds_addr | global_addr | type=2
        g0[0] = 1u;
        g0[1] = lds_off;
        g0[2] = (unsigned)ga;
        g0[3] = (unsigned)((ga >> 32) & 0x1FFFFFFu) | (2u << 30);
        // D# group1: data_size=2B; tensor_dim0=K, tensor_dim1=rows,
        //            tile_dim0=K, tile_dim1=64, tensor_dim0_stride=K
        g1[0] = (int)(1u << 16);                              // data_size=1 (2B)
        g1[1] = (int)((unsigned)(K & 0xFFFF) << 16);          // tensor_dim0 lo16
        g1[2] = (int)(((unsigned)K >> 16) |
                      ((unsigned)(rows & 0xFFFF) << 16));     // dim0 hi | dim1 lo
        g1[3] = (int)((unsigned)(K & 0xFFFF) << 16);          // tile_dim0 = K
        g1[4] = 64;                                           // tile_dim1 = 64 rows
        g1[5] = K;                                            // dim0 stride lo32
        g1[6] = 0;
        g1[7] = 0;
        __builtin_amdgcn_tensor_load_to_lds(g0, g1, g2, g3, g4, 0);
        __builtin_amdgcn_s_wait_tensorcnt(0);
    }
#else
    for (int idx = tid * 8; idx < 64 * K; idx += 256 * 8) {
        const int n = idx / K, c = idx - n * K;
        u32x4 wv = {0u, 0u, 0u, 0u};
        if (bn + n < N) wv = *(const u32x4*)(W + (size_t)(bn + n) * K + c);
        *(u32x4*)(Ws + idx) = wv;
    }
#endif
}

// ---------------------------------------------------------------------------
// WMMA GEMM:  C[M,N] = act( A[M,K] @ W[N,K]^T + bias )
// A,W bf16; C f32 (mode 0) / bf16 linear (mode 1) / bf16 softplus (mode 2).
// Block = 256 thr = 8 waves; block tile 64(M) x 64(N); W panel resident in
// LDS for the whole K loop; each wave owns two M-subtiles sharing one B
// fragment -> 2 x v_wmma per K-step per wave.
// M % 64 == 0, K % 32 == 0, K <= 256; N guarded.
// ---------------------------------------------------------------------------
__global__ __launch_bounds__(256) void gemm_bf16_kernel(
    const bf16* __restrict__ A, int lda,
    const bf16* __restrict__ W, int N, int K,
    const float* __restrict__ bias,
    float* __restrict__ Cf, bf16* __restrict__ Cb, int ldc, int mode)
{
    __shared__ bf16 Ws[64 * 256];   // resident weight panel (<= 32 KB)
    __shared__ bf16 As[64 * 32];
    const int tid  = threadIdx.x;
    const int bm   = blockIdx.y * 64;
    const int bn   = blockIdx.x * 64;
    const int lane = tid & 31;
    const int wave = tid >> 5;
    const int wn   = wave & 3;          // N subtile
    const int wmg  = wave >> 2;         // 0..1 -> M subtiles 2*wmg, 2*wmg+1
    const int mrow = lane & 15;
    const bool hi  = lane >= 16;

    stage_w_panel(W, N, K, bn, Ws, tid);
    __syncthreads();

    v8f acc0 = {0.f, 0.f, 0.f, 0.f, 0.f, 0.f, 0.f, 0.f};
    v8f acc1 = {0.f, 0.f, 0.f, 0.f, 0.f, 0.f, 0.f, 0.f};

    for (int kb = 0; kb < K; kb += 32) {
        // A tile 64x32: 16B per thread
        {
            const int r = tid >> 2, c = (tid & 3) << 3;
            *(u32x4*)(As + r * 32 + c) =
                *(const u32x4*)(A + (size_t)(bm + r) * lda + kb + c);
        }
        __syncthreads();

        FragBf a0, a1, b;
        const bf16* ap0 = As + (wmg * 32 + mrow) * 32;
        const bf16* ap1 = ap0 + 16 * 32;
        a0.q[0] = *(const u32x4*)(ap0 + (hi ? 8 : 0));
        a0.q[1] = *(const u32x4*)(ap0 + (hi ? 24 : 16));
        a1.q[0] = *(const u32x4*)(ap1 + (hi ? 8 : 0));
        a1.q[1] = *(const u32x4*)(ap1 + (hi ? 24 : 16));
        const bf16* bp = Ws + (size_t)(wn * 16 + mrow) * K + kb + (hi ? 8 : 0);
        b.q[0] = *(const u32x4*)(bp);
        b.q[1] = *(const u32x4*)(bp + 16);

        acc0 = __builtin_amdgcn_wmma_f32_16x16x32_bf16(
            false, a0.v, false, b.v, (short)0, acc0, false, false);
        acc1 = __builtin_amdgcn_wmma_f32_16x16x32_bf16(
            false, a1.v, false, b.v, (short)0, acc1, false, false);
        __syncthreads();
    }

    const int col = bn + wn * 16 + mrow;
    if (col < N) {
        const float bv = bias ? bias[col] : 0.f;
#pragma unroll
        for (int r = 0; r < 8; ++r) {
            const int row0 = bm + wmg * 32 + (hi ? r + 8 : r);
            const int row1 = row0 + 16;
            const float v0 = acc0[r] + bv;
            const float v1 = acc1[r] + bv;
            const size_t o0 = (size_t)row0 * ldc + col;
            const size_t o1 = (size_t)row1 * ldc + col;
            if (mode == 0)      { Cf[o0] = v0;               Cf[o1] = v1; }
            else if (mode == 1) { Cb[o0] = (bf16)v0;         Cb[o1] = (bf16)v1; }
            else                { Cb[o0] = (bf16)splus(v0);  Cb[o1] = (bf16)splus(v1); }
        }
    }
}

// ---------------------------------------------------------------------------
// Poisson log-likelihood: WMMA GEMM (S*B*T x 128 x 32) with fused
// ell = x*lr - exp(lr) - lgamma(x+1) epilogue; block reduce + atomicAdd.
// ---------------------------------------------------------------------------
__global__ __launch_bounds__(256) void poisson_ell_kernel(
    const bf16* __restrict__ Zbf,      // (S*B*T, 32) bf16
    const bf16* __restrict__ Wt,       // (128, 32) = C_out^T bf16
    const float* __restrict__ dvec,    // (128)
    const float* __restrict__ xs,      // (B,T,D) f32
    float log_dt, float* __restrict__ ell_acc)
{
    __shared__ bf16 As[32 * 32];
    __shared__ bf16 Ws[64 * 32];
    __shared__ float red[256];
    const int tid  = threadIdx.x;
    const int bm   = blockIdx.y * 32;
    const int bn   = blockIdx.x * 64;
    const int lane = tid & 31;
    const int wave = tid >> 5;
    const int wm   = wave >> 2;
    const int wn   = wave & 3;
    const int mrow = lane & 15;
    const bool hi  = lane >= 16;

    v8f acc = {0.f, 0.f, 0.f, 0.f, 0.f, 0.f, 0.f, 0.f};
    {   // K = 32, single step
        const int r = tid >> 3, c = (tid & 7) << 2;
        *(u32x2*)(As + r * 32 + c) = *(const u32x2*)(Zbf + (size_t)(bm + r) * 32 + c);
        const int n = tid >> 2, c8 = (tid & 3) << 3;
        *(u32x4*)(Ws + n * 32 + c8) = *(const u32x4*)(Wt + (size_t)(bn + n) * 32 + c8);
        __syncthreads();

        FragBf a, b;
        const bf16* ap = As + (wm * 16 + mrow) * 32;
        a.q[0] = *(const u32x4*)(ap + (hi ? 8 : 0));
        a.q[1] = *(const u32x4*)(ap + (hi ? 24 : 16));
        const bf16* bp = Ws + (wn * 16 + mrow) * 32 + (hi ? 8 : 0);
        b.q[0] = *(const u32x4*)(bp);
        b.q[1] = *(const u32x4*)(bp + 16);
        acc = __builtin_amdgcn_wmma_f32_16x16x32_bf16(
            false, a.v, false, b.v, (short)0, acc, false, false);
        __syncthreads();
    }

    const int col = bn + wn * 16 + mrow;     // d in [0,128): always valid
    float local = 0.f;
#pragma unroll
    for (int r = 0; r < 8; ++r) {
        const int row = bm + wm * 16 + (hi ? r + 8 : r);   // (s*B+b)*T + t
        const int bt  = row % (B_ * T_);                   // b*T + t
        const float lr = acc[r] + dvec[col] + log_dt;
        const float x  = xs[(size_t)bt * D_ + col];
        local += x * lr - __expf(lr) - lgammaf(x + 1.f);
    }
    red[tid] = local;
    __syncthreads();
    for (int s = 128; s > 0; s >>= 1) {
        if (tid < s) red[tid] += red[tid + s];
        __syncthreads();
    }
    if (tid == 0) atomicAdd(ell_acc, red[0]);
}

// ------------------------------ elementwise --------------------------------
__global__ void tobf16_kernel(const float* __restrict__ s, bf16* __restrict__ d, int n) {
    int i = blockIdx.x * blockDim.x + threadIdx.x;
    if (i < n) d[i] = (bf16)s[i];
}
__global__ void transpose_cout_kernel(const float* __restrict__ Cout, bf16* __restrict__ Wt) {
    int i = blockIdx.x * blockDim.x + threadIdx.x;   // i = d*32 + l
    if (i < D_ * L_) { int d = i >> 5, l = i & 31; Wt[i] = (bf16)Cout[l * D_ + d]; }
}
__global__ void init_state_kernel(float* h_f32, bf16* h_bf, float* path_acc,
                                  float* ell_acc, float* kl_acc) {
    int i = blockIdx.x * blockDim.x + threadIdx.x;   // 65536 threads
    if (i < B_ * H_) { h_f32[i] = 0.f; h_bf[i] = (bf16)0.f; }
    if (i < SB_) path_acc[i] = 0.f;
    if (i == 0) { ell_acc[0] = 0.f; kl_acc[0] = 0.f; }
}
__global__ void gru_gates_kernel(const float* __restrict__ gi, const float* __restrict__ gh,
                                 float* __restrict__ h_f32, bf16* __restrict__ h_bf,
                                 bf16* __restrict__ hs_t) {
    int i = blockIdx.x * blockDim.x + threadIdx.x;   // 65536
    int b = i >> 8, j = i & 255;
    const float* gib = gi + (size_t)b * 768;
    const float* ghb = gh + (size_t)b * 768;
    float r = sigm(gib[j] + ghb[j]);
    float u = sigm(gib[256 + j] + ghb[256 + j]);
    float n = tanhf(gib[512 + j] + r * ghb[512 + j]);
    float hn = (1.f - u) * n + u * h_f32[i];
    h_f32[i] = hn; h_bf[i] = (bf16)hn; hs_t[i] = (bf16)hn;
}
__global__ void z0_init_kernel(const float* __restrict__ q, const float* __restrict__ eps0,
                               float* __restrict__ z_f32, bf16* __restrict__ fin,
                               float* __restrict__ zs_f, bf16* __restrict__ zs_b) {
    int i = blockIdx.x * blockDim.x + threadIdx.x;   // SB*L = 131072
    int m = i >> 5, l = i & 31, b = m & (B_ - 1);
    float z = q[b * 64 + l] + __expf(q[b * 64 + 32 + l]) * eps0[i];
    z_f32[i] = z;
    fin[(size_t)m * 96 + l] = (bf16)z;
    size_t o = ((size_t)m * T_) * L_ + l;            // t = 0
    zs_f[o] = z; zs_b[o] = (bf16)z;
}
__global__ void pack_ctx_kernel(const bf16* __restrict__ ctx_t, bf16* __restrict__ fin) {
    int i = blockIdx.x * blockDim.x + threadIdx.x;   // SB*C = 262144
    int m = i >> 6, c = i & 63, b = m & (B_ - 1);
    fin[(size_t)m * 96 + 32 + c] = ctx_t[b * C_ + c];
}
__global__ void gfn_kernel(const float* __restrict__ z_f32,
                           const float* __restrict__ gW1, const float* __restrict__ gb1,
                           const float* __restrict__ gW2, const float* __restrict__ gb2,
                           float* __restrict__ gv) {
    int i = blockIdx.x * blockDim.x + threadIdx.x;   // SB*L
    int l = i & 31;
    float z = z_f32[i], acc = 0.f;
    const float* w1 = gW1 + l * H_;
    const float* b1 = gb1 + l * H_;
    const float* w2 = gW2 + l * H_;
#pragma unroll 4
    for (int h = 0; h < H_; ++h) acc += splus(z * w1[h] + b1[h]) * w2[h];
    gv[i] = sigm(acc + gb2[l]) + 0.01f;
}
__global__ void euler_kernel(const float* __restrict__ fv, const float* __restrict__ hv,
                             const float* __restrict__ gv, const float* __restrict__ dWk,
                             float* __restrict__ z_f32, bf16* __restrict__ fin,
                             float* __restrict__ zs_f, bf16* __restrict__ zs_b,
                             float* __restrict__ path_acc, int t, float sq_dt) {
    int m = blockIdx.x * blockDim.x + threadIdx.x;   // SB threads
    if (m >= SB_) return;
    float ssum = 0.f;
#pragma unroll
    for (int l = 0; l < L_; ++l) {
        int i = m * L_ + l;
        float f = fv[i], h = hv[i], g = gv[i], z = z_f32[i];
        float u = (f - h) / g;
        ssum += u * u;
        float zn = z + f * DT_ + g * (sq_dt * dWk[i]);
        z_f32[i] = zn;
        fin[(size_t)m * 96 + l] = (bf16)zn;
        size_t o = ((size_t)m * T_ + t) * L_ + l;
        zs_f[o] = zn; zs_b[o] = (bf16)zn;
    }
    path_acc[m] += 0.5f * ssum * DT_;
}
__global__ void kl_kernel(const float* __restrict__ q, const float* __restrict__ pm,
                          const float* __restrict__ pl, float* __restrict__ kl_acc) {
    __shared__ float red[256];
    int i = blockIdx.x * blockDim.x + threadIdx.x;   // B*L = 8192
    int b = i >> 5, l = i & 31;
    float qm = q[b * 64 + l], ql = q[b * 64 + 32 + l];
    float dmu = qm - pm[l];
    float kl = pl[l] - ql + (__expf(2.f * ql) + dmu * dmu) / (2.f * __expf(2.f * pl[l])) - 0.5f;
    red[threadIdx.x] = kl;
    __syncthreads();
    for (int s = 128; s > 0; s >>= 1) {
        if (threadIdx.x < s) red[threadIdx.x] += red[threadIdx.x + s];
        __syncthreads();
    }
    if (threadIdx.x == 0) atomicAdd(kl_acc, red[0]);
}
__global__ void stats_kernel(const float* __restrict__ zs_f, float* __restrict__ m_out,
                             float* __restrict__ p_out) {
    int i = blockIdx.x * blockDim.x + threadIdx.x;   // B*T*L = 1048576
    float s1 = 0.f, s2 = 0.f;
#pragma unroll
    for (int s = 0; s < S_; ++s) {
        float v = zs_f[(size_t)s * (B_ * T_ * L_) + i];
        s1 += v; s2 += v * v;
    }
    float mean = s1 * (1.f / S_);
    m_out[i] = mean;
    p_out[i] = (s2 - S_ * mean * mean) * (1.f / (S_ - 1));
}
__global__ void finalize_kernel(const float* __restrict__ path_acc,
                                const float* __restrict__ ell_acc,
                                const float* __restrict__ kl_acc, float* __restrict__ loss) {
    __shared__ float red[256];
    float s = 0.f;
    for (int i = threadIdx.x; i < SB_; i += 256) s += path_acc[i];
    red[threadIdx.x] = s;
    __syncthreads();
    for (int k = 128; k > 0; k >>= 1) {
        if (threadIdx.x < k) red[threadIdx.x] += red[threadIdx.x + k];
        __syncthreads();
    }
    if (threadIdx.x == 0)
        loss[0] = -(ell_acc[0] / (float)SB_) + kl_acc[0] / (float)B_ + red[0] / (float)SB_;
}

// ------------------------------ host side ----------------------------------
static inline void launch_gemm(hipStream_t st, const bf16* A, int lda, const bf16* W,
                               int M, int N, int K, const float* bias,
                               float* Cf, bf16* Cb, int ldc, int mode) {
    dim3 g((N + 63) / 64, M / 64), b(256);
    gemm_bf16_kernel<<<g, b, 0, st>>>(A, lda, W, N, K, bias, Cf, Cb, ldc, mode);
}

extern "C" void kernel_launch(void* const* d_in, const int* in_sizes, int n_in,
                              void* d_out, int out_size, void* d_ws, size_t ws_size,
                              hipStream_t stream) {
    const float* xs_pre  = (const float*)d_in[0];
    const float* eps0    = (const float*)d_in[1];
    const float* dW      = (const float*)d_in[2];
    const float* Wih     = (const float*)d_in[3];
    const float* Whh     = (const float*)d_in[4];
    const float* bih     = (const float*)d_in[5];
    const float* bhh     = (const float*)d_in[6];
    const float* encW    = (const float*)d_in[7];
    const float* encb    = (const float*)d_in[8];
    const float* qz0W    = (const float*)d_in[9];
    const float* qz0b    = (const float*)d_in[10];
    const float* fW1 = (const float*)d_in[11]; const float* fb1 = (const float*)d_in[12];
    const float* fW2 = (const float*)d_in[13]; const float* fb2 = (const float*)d_in[14];
    const float* fW3 = (const float*)d_in[15]; const float* fb3 = (const float*)d_in[16];
    const float* hW1 = (const float*)d_in[17]; const float* hb1 = (const float*)d_in[18];
    const float* hW2 = (const float*)d_in[19]; const float* hb2 = (const float*)d_in[20];
    const float* hW3 = (const float*)d_in[21]; const float* hb3 = (const float*)d_in[22];
    const float* gW1 = (const float*)d_in[23]; const float* gb1 = (const float*)d_in[24];
    const float* gW2 = (const float*)d_in[25]; const float* gb2 = (const float*)d_in[26];
    const float* pz0m = (const float*)d_in[27]; const float* pz0l = (const float*)d_in[28];
    const float* Cout = (const float*)d_in[29]; const float* dvec = (const float*)d_in[30];

    float* out      = (float*)d_out;
    float* zs_f     = out + 1;                              // (S,B,T,L)
    float* m_out    = zs_f + (size_t)S_ * B_ * T_ * L_;
    float* p_out    = m_out + (size_t)B_ * T_ * L_;

    // ---- workspace bump allocator ----
    char* wsb = (char*)d_ws;
    size_t off = 0;
    auto alloc = [&](size_t bytes) -> void* {
        void* p = wsb + off;
        off = (off + bytes + 255) & ~(size_t)255;
        return p;
    };
    bf16* xs_bf  = (bf16*)alloc((size_t)B_ * T_ * D_ * 2);
    bf16* Wih_b  = (bf16*)alloc(768 * 128 * 2);
    bf16* Whh_b  = (bf16*)alloc(768 * 256 * 2);
    bf16* encW_b = (bf16*)alloc(C_ * H_ * 2);
    bf16* qz0W_b = (bf16*)alloc(64 * 64 * 2);
    bf16* fW1_b  = (bf16*)alloc(256 * 96 * 2);
    bf16* fW2_b  = (bf16*)alloc(256 * 256 * 2);
    bf16* fW3_b  = (bf16*)alloc(32 * 256 * 2);
    bf16* hW1_b  = (bf16*)alloc(256 * 32 * 2);
    bf16* hW2_b  = (bf16*)alloc(256 * 256 * 2);
    bf16* hW3_b  = (bf16*)alloc(32 * 256 * 2);
    bf16* CoutT  = (bf16*)alloc(D_ * L_ * 2);
    bf16* hs_bf  = (bf16*)alloc((size_t)T_ * B_ * H_ * 2);
    bf16* ctx_bf = (bf16*)alloc((size_t)T_ * B_ * C_ * 2);
    float* h_f32 = (float*)alloc((size_t)B_ * H_ * 4);
    bf16*  h_bf  = (bf16*)alloc((size_t)B_ * H_ * 2);
    float* gi    = (float*)alloc((size_t)B_ * 768 * 4);
    float* gh    = (float*)alloc((size_t)B_ * 768 * 4);
    float* q     = (float*)alloc((size_t)B_ * 64 * 4);
    bf16*  fin   = (bf16*)alloc((size_t)SB_ * 96 * 2);
    bf16*  a_bf  = (bf16*)alloc((size_t)SB_ * 256 * 2);
    bf16*  b_bf  = (bf16*)alloc((size_t)SB_ * 256 * 2);
    float* fv    = (float*)alloc((size_t)SB_ * L_ * 4);
    float* hv    = (float*)alloc((size_t)SB_ * L_ * 4);
    float* gv    = (float*)alloc((size_t)SB_ * L_ * 4);
    float* z_f32 = (float*)alloc((size_t)SB_ * L_ * 4);
    bf16*  zs_b  = (bf16*)alloc((size_t)S_ * B_ * T_ * L_ * 2);
    float* path_acc = (float*)alloc(SB_ * 4);
    float* ell_acc  = (float*)alloc(256);
    float* kl_acc   = (float*)alloc(256);
    (void)ws_size; (void)in_sizes; (void)n_in; (void)out_size;

    auto conv = [&](const float* s, bf16* d, int n) {
        tobf16_kernel<<<(n + 255) / 256, 256, 0, stream>>>(s, d, n);
    };
    conv(xs_pre, xs_bf, B_ * T_ * D_);
    conv(Wih, Wih_b, 768 * 128);   conv(Whh, Whh_b, 768 * 256);
    conv(encW, encW_b, C_ * H_);   conv(qz0W, qz0W_b, 64 * 64);
    conv(fW1, fW1_b, 256 * 96);    conv(fW2, fW2_b, 256 * 256);  conv(fW3, fW3_b, 32 * 256);
    conv(hW1, hW1_b, 256 * 32);    conv(hW2, hW2_b, 256 * 256);  conv(hW3, hW3_b, 32 * 256);
    transpose_cout_kernel<<<(D_ * L_ + 255) / 256, 256, 0, stream>>>(Cout, CoutT);
    init_state_kernel<<<256, 256, 0, stream>>>(h_f32, h_bf, path_acc, ell_acc, kl_acc);

    // ---- GRU over reversed time; store hidden at t = T-1-k (pre-flipped) ----
    for (int k = 0; k < T_; ++k) {
        const int t = T_ - 1 - k;
        launch_gemm(stream, xs_bf + (size_t)t * D_, T_ * D_, Wih_b,
                    B_, 768, 128, bih, gi, nullptr, 768, 0);
        launch_gemm(stream, h_bf, H_, Whh_b, B_, 768, H_, bhh, gh, nullptr, 768, 0);
        gru_gates_kernel<<<256, 256, 0, stream>>>(gi, gh, h_f32, h_bf,
                                                  hs_bf + (size_t)t * B_ * H_);
    }
    // ctx = hs @ encW^T + encb  (already in flipped-back time order), bf16 out
    launch_gemm(stream, hs_bf, H_, encW_b, T_ * B_, C_, H_, encb,
                nullptr, ctx_bf, C_, 1);
    // q = ctx[0] @ qz0W^T + qz0b (f32)
    launch_gemm(stream, ctx_bf, C_, qz0W_b, B_, 64, 64, qz0b, q, nullptr, 64, 0);
    z0_init_kernel<<<(SB_ * L_ + 255) / 256, 256, 0, stream>>>(q, eps0, z_f32, fin,
                                                               zs_f, zs_b);

    // ---- Euler-Maruyama scan ----
    const float sq_dt = sqrtf(DT_);
    for (int k = 0; k < T_ - 1; ++k) {
        pack_ctx_kernel<<<(SB_ * C_ + 255) / 256, 256, 0, stream>>>(
            ctx_bf + (size_t)(k + 1) * B_ * C_, fin);
        // f_net: softplus fused into GEMM epilogue (mode 2)
        launch_gemm(stream, fin, 96, fW1_b, SB_, 256, 96, fb1, nullptr, a_bf, 256, 2);
        launch_gemm(stream, a_bf, 256, fW2_b, SB_, 256, 256, fb2, nullptr, b_bf, 256, 2);
        launch_gemm(stream, b_bf, 256, fW3_b, SB_, 32, 256, fb3, fv, nullptr, 32, 0);
        // h_net (reads z = first 32 cols of fin)
        launch_gemm(stream, fin, 96, hW1_b, SB_, 256, 32, hb1, nullptr, a_bf, 256, 2);
        launch_gemm(stream, a_bf, 256, hW2_b, SB_, 256, 256, hb2, nullptr, b_bf, 256, 2);
        launch_gemm(stream, b_bf, 256, hW3_b, SB_, 32, 256, hb3, hv, nullptr, 32, 0);
        // diagonal g net
        gfn_kernel<<<(SB_ * L_ + 255) / 256, 256, 0, stream>>>(z_f32, gW1, gb1, gW2, gb2, gv);
        euler_kernel<<<(SB_ + 255) / 256, 256, 0, stream>>>(
            fv, hv, gv, dW + (size_t)k * SB_ * L_, z_f32, fin, zs_f, zs_b,
            path_acc, k + 1, sq_dt);
    }

    // ---- Poisson likelihood (fused WMMA GEMM + ell reduction) ----
    {
        dim3 g((D_ + 63) / 64, (S_ * B_ * T_) / 32), b(256);
        poisson_ell_kernel<<<g, b, 0, stream>>>(zs_b, CoutT, dvec, xs_pre,
                                                logf(DT_), ell_acc);
    }
    kl_kernel<<<(B_ * L_ + 255) / 256, 256, 0, stream>>>(q, pz0m, pz0l, kl_acc);
    stats_kernel<<<(B_ * T_ * L_ + 255) / 256, 256, 0, stream>>>(zs_f, m_out, p_out);
    finalize_kernel<<<1, 256, 0, stream>>>(path_acc, ell_acc, kl_acc, out);
}
